// CustomSelfAttention_64991445123214
// MI455X (gfx1250) — matrix operand order repaired and checked by reading, not compiled, not run on hardware
//
#include <hip/hip_runtime.h>
#include <hip/hip_bf16.h>

// Problem constants (reference: B=2, S=2048, D=1024, H=16, dk=64)
#define S_LEN   2048
#define D_MODEL 1024
#define NHEAD   16
#define DK      64
#define BATCH   2
#define PROW    2049   // padded LDS row stride (floats) to avoid bank conflicts

typedef __attribute__((ext_vector_type(16))) __bf16 v16bf;
typedef __attribute__((ext_vector_type(8)))  float  v8f;

union V16BF {
    v16bf v;
    unsigned int u[8];
    uint4 q[2];
};

// K index of element pair (2*vv, 2*vv+1) inside a 16-bit A fragment (16x32),
// contiguous order matching B: lane half h holds K 16*h .. 16*h+15.
__device__ __forceinline__ int kbase(int vv, int half) {
    return 16 * half + 2 * vv;
}

// Load A fragment from a row-major bf16 row pointer (row of matrix element M = lane&15).
__device__ __forceinline__ v16bf load_a_rowmajor(const unsigned short* row, int lane) {
    const int half = lane >> 4;
    V16BF a;
#pragma unroll
    for (int vv = 0; vv < 8; ++vv)
        a.u[vv] = *(const unsigned int*)(row + kbase(vv, half));
    return a.v;
}

// Load B fragment for NT GEMM (B element (k, n) = M2[n][k], row-major M2):
// lane covers column n = lane&15, K range 16*(lane>>4) .. +15 contiguous.
__device__ __forceinline__ v16bf load_b_nt(const unsigned short* row_at_khalf) {
    V16BF b;
    b.q[0] = *(const uint4*)(row_at_khalf);
    b.q[1] = *(const uint4*)(row_at_khalf + 8);
    return b.v;
}

__device__ __forceinline__ unsigned int pack2_bf16(float lo, float hi) {
    __bf16 a = (__bf16)lo, b = (__bf16)hi;
    unsigned short ua = __builtin_bit_cast(unsigned short, a);
    unsigned short ub = __builtin_bit_cast(unsigned short, b);
    return (unsigned int)ua | ((unsigned int)ub << 16);
}

__device__ __forceinline__ unsigned short to_bf16_bits(float f) {
    __bf16 h = (__bf16)f;
    return __builtin_bit_cast(unsigned short, h);
}

// ---------------------------------------------------------------------------
// 1) fp32 -> bf16 conversion, 4 elements per thread
// ---------------------------------------------------------------------------
__global__ void cvt_bf16x4_kernel(const float4* __restrict__ in,
                                  uint2* __restrict__ out, int n4) {
    int i = blockIdx.x * blockDim.x + threadIdx.x;
    if (i < n4) {
        float4 f = in[i];
        uint2 r;
        r.x = pack2_bf16(f.x, f.y);
        r.y = pack2_bf16(f.z, f.w);
        out[i] = r;
    }
}

// ---------------------------------------------------------------------------
// 2) QKV projection: Y = X @ W^T + b  (bf16 in, fp32 acc, bf16 out)
//    One wave per 16x64 output tile (4 accumulators, A fragment reused 4x).
//    v_transposed==0: out[b][h][s][d]   (Q, K layout)
//    v_transposed==1: out[b][h][d][s]   (V transposed for PV B-fragments)
// ---------------------------------------------------------------------------
__global__ void __launch_bounds__(32)
proj_qkv_kernel(const unsigned short* __restrict__ X,
                const unsigned short* __restrict__ W,
                const float* __restrict__ bias,
                unsigned short* __restrict__ out,
                int v_transposed) {
    const int lane = threadIdx.x;
    const int half = lane >> 4;
    const int lr   = lane & 15;
    const int m0   = blockIdx.x * 16;   // row tile over B*S
    const int n0   = blockIdx.y * 64;   // col tile over D (4 x 16)

    const v8f zero = {0.f, 0.f, 0.f, 0.f, 0.f, 0.f, 0.f, 0.f};
    v8f acc[4] = {zero, zero, zero, zero};

    const unsigned short* arow = X + (size_t)(m0 + lr) * D_MODEL;
    const unsigned short* brow[4];
#pragma unroll
    for (int nb = 0; nb < 4; ++nb)
        brow[nb] = W + (size_t)(n0 + nb * 16 + lr) * D_MODEL + 16 * half;

#pragma unroll 2
    for (int k0 = 0; k0 < D_MODEL; k0 += 32) {
        v16bf a = load_a_rowmajor(arow + k0, lane);
#pragma unroll
        for (int nb = 0; nb < 4; ++nb) {
            v16bf b = load_b_nt(brow[nb] + k0);
            acc[nb] = __builtin_amdgcn_wmma_f32_16x16x32_bf16(false, a, false, b,
                                                              (short)0, acc[nb], false, false);
        }
    }

#pragma unroll
    for (int nb = 0; nb < 4; ++nb) {
        const int n  = n0 + nb * 16 + lr;
        const int h  = n >> 6;
        const int dd = n & (DK - 1);
        const float bn = bias[n];
#pragma unroll
        for (int vv = 0; vv < 8; ++vv) {
            const int m     = vv + 8 * half;
            const int sflat = m0 + m;
            const int bb    = sflat >> 11;          // / S_LEN
            const int ss    = sflat & (S_LEN - 1);
            const float val = acc[nb][vv] + bn;
            size_t idx;
            if (v_transposed)
                idx = ((size_t)(bb * NHEAD + h) * DK + dd) * S_LEN + ss;
            else
                idx = ((size_t)(bb * NHEAD + h) * S_LEN + ss) * DK + dd;
            out[idx] = to_bf16_bits(val);
        }
    }
}

// ---------------------------------------------------------------------------
// 3) Fused attention, 4 waves per workgroup sharing one 16 x S score buffer.
//    Phase A: waves split key tiles; WMMA bf16 scores -> LDS (causal -inf).
//    Phase B: waves split rows 4-way; exact fp32 softmax; single streaming
//             p_attn write (the roofline-dominant 537 MB HBM store).
//    Phase C: waves split dk 4-way; PV via WMMA (P from LDS, V^T from L2).
// ---------------------------------------------------------------------------
__global__ void __launch_bounds__(128)
attn_kernel(const unsigned short* __restrict__ Qh,
            const unsigned short* __restrict__ Kh,
            const unsigned short* __restrict__ Vt,
            float* __restrict__ p_attn,
            unsigned short* __restrict__ ctx) {
    extern __shared__ float P[];          // [16][PROW]
    const int tid  = threadIdx.x;
    const int w    = tid >> 5;            // wave id 0..3
    const int lane = tid & 31;
    const int half = lane >> 4;
    const int lr   = lane & 15;
    const int q0   = blockIdx.x * 16;
    const int h    = blockIdx.y;
    const int b    = blockIdx.z;
    const size_t bh = (size_t)b * NHEAD + h;

    const unsigned short* Qb = Qh + bh * (size_t)S_LEN * DK;
    const unsigned short* Kb = Kh + bh * (size_t)S_LEN * DK;
    const unsigned short* Vb = Vt + bh * (size_t)DK * S_LEN;

    // Q tile A-fragments (constant over the key loop): K-chunks 0 and 32 of dk=64
    const unsigned short* qrow = Qb + (size_t)(q0 + lr) * DK;
    const v16bf a0 = load_a_rowmajor(qrow + 0, lane);
    const v16bf a1 = load_a_rowmajor(qrow + 32, lane);

    const float scale = 0.03125f;  // 1/sqrt(D)=1/32 (faithful to reference's sqrt(d_model))
    const v8f zero = {0.f, 0.f, 0.f, 0.f, 0.f, 0.f, 0.f, 0.f};

    // ---- Phase A: scores into LDS, waves stride over key tiles ----
    for (int j0 = 16 * w; j0 <= q0; j0 += 64) {
        const unsigned short* krow = Kb + (size_t)(j0 + lr) * DK + 16 * half;
        __builtin_prefetch(krow + 64 * DK, 0, 0);   // this wave's next key tile
        v8f s = zero;
        v16bf b0 = load_b_nt(krow);
        v16bf b1 = load_b_nt(krow + 32);
        s = __builtin_amdgcn_wmma_f32_16x16x32_bf16(false, a0, false, b0, (short)0, s, false, false);
        s = __builtin_amdgcn_wmma_f32_16x16x32_bf16(false, a1, false, b1, (short)0, s, false, false);
#pragma unroll
        for (int vv = 0; vv < 8; ++vv) {
            const int m   = vv + 8 * half;
            const int col = j0 + lr;
            float val = s[vv] * scale;
            if (col > q0 + m) val = -__builtin_inff();
            P[m * PROW + col] = val;
        }
    }
    __syncthreads();

    // ---- Phase B: softmax; wave w owns rows 4w..4w+3 ----
    const int kmax = (q0 + 16 + 31) & ~31;   // PV key range, rounded up to 32
    for (int rr = 0; rr < 4; ++rr) {
        const int r     = w * 4 + rr;
        const int limit = q0 + r + 1;
        float mx = -__builtin_inff();
        for (int col = lane; col < limit; col += 32)
            mx = fmaxf(mx, P[r * PROW + col]);
#pragma unroll
        for (int off = 16; off; off >>= 1)
            mx = fmaxf(mx, __shfl_xor(mx, off, 32));
        float sum = 0.f;
        for (int col = lane; col < limit; col += 32) {
            const float e = __expf(P[r * PROW + col] - mx);
            P[r * PROW + col] = e;
            sum += e;
        }
#pragma unroll
        for (int off = 16; off; off >>= 1)
            sum += __shfl_xor(sum, off, 32);
        const float inv = 1.f / sum;
        float* prow = p_attn + (bh * S_LEN + (size_t)(q0 + r)) * S_LEN;
        for (int col = lane; col < S_LEN; col += 32) {
            const float pv = (col < limit) ? P[r * PROW + col] * inv : 0.f;
            prow[col] = pv;                       // dominant HBM write, streamed once
            if (col < kmax) P[r * PROW + col] = pv;  // zero masked tail for PV tiles
        }
    }
    __syncthreads();

    // ---- Phase C: PV; wave w computes dk columns [16w, 16w+16) ----
    v8f acc = zero;
    const unsigned short* vbase = Vb + (size_t)(w * 16 + lr) * S_LEN + 16 * half;
    for (int kc = 0; kc < kmax; kc += 32) {
        // A fragment from LDS probabilities (fp32 -> bf16 pairs)
        V16BF pa;
        const float* prow = P + lr * PROW + kc;
#pragma unroll
        for (int vv = 0; vv < 8; ++vv) {
            const int kk = kbase(vv, half);
            pa.u[vv] = pack2_bf16(prow[kk], prow[kk + 1]);
        }
        v16bf bv = load_b_nt(vbase + kc);
        acc = __builtin_amdgcn_wmma_f32_16x16x32_bf16(false, pa.v, false, bv,
                                                      (short)0, acc, false, false);
    }

    // ctx layout [B][S][D], bf16 (input to output projection)
#pragma unroll
    for (int vv = 0; vv < 8; ++vv) {
        const int m  = vv + 8 * half;
        const int dd = w * 16 + lr;
        ctx[((size_t)(b * S_LEN + q0 + m)) * D_MODEL + h * DK + dd] =
            to_bf16_bits(acc[vv]);
    }
}

// ---------------------------------------------------------------------------
// 4) Output projection: out = ctx @ Wo^T + bo (fp32 result), 16x64 per wave
// ---------------------------------------------------------------------------
__global__ void __launch_bounds__(32)
proj_out_kernel(const unsigned short* __restrict__ X,
                const unsigned short* __restrict__ W,
                const float* __restrict__ bias,
                float* __restrict__ out) {
    const int lane = threadIdx.x;
    const int half = lane >> 4;
    const int lr   = lane & 15;
    const int m0   = blockIdx.x * 16;
    const int n0   = blockIdx.y * 64;

    const v8f zero = {0.f, 0.f, 0.f, 0.f, 0.f, 0.f, 0.f, 0.f};
    v8f acc[4] = {zero, zero, zero, zero};

    const unsigned short* arow = X + (size_t)(m0 + lr) * D_MODEL;
    const unsigned short* brow[4];
#pragma unroll
    for (int nb = 0; nb < 4; ++nb)
        brow[nb] = W + (size_t)(n0 + nb * 16 + lr) * D_MODEL + 16 * half;

#pragma unroll 2
    for (int k0 = 0; k0 < D_MODEL; k0 += 32) {
        v16bf a = load_a_rowmajor(arow + k0, lane);
#pragma unroll
        for (int nb = 0; nb < 4; ++nb) {
            v16bf b = load_b_nt(brow[nb] + k0);
            acc[nb] = __builtin_amdgcn_wmma_f32_16x16x32_bf16(false, a, false, b,
                                                              (short)0, acc[nb], false, false);
        }
    }

#pragma unroll
    for (int nb = 0; nb < 4; ++nb) {
        const int n  = n0 + nb * 16 + lr;
        const float bn = bias[n];
#pragma unroll
        for (int vv = 0; vv < 8; ++vv) {
            const int m = vv + 8 * half;
            out[(size_t)(m0 + m) * D_MODEL + n] = acc[nb][vv] + bn;
        }
    }
}

// ---------------------------------------------------------------------------
// Launch
// ---------------------------------------------------------------------------
extern "C" void kernel_launch(void* const* d_in, const int* in_sizes, int n_in,
                              void* d_out, int out_size, void* d_ws, size_t ws_size,
                              hipStream_t stream) {
    (void)in_sizes; (void)n_in; (void)out_size; (void)ws_size;

    const float* x_q = (const float*)d_in[0];
    const float* x_k = (const float*)d_in[1];
    const float* x_v = (const float*)d_in[2];
    const float* Wq  = (const float*)d_in[3];
    const float* bq  = (const float*)d_in[4];
    const float* Wk  = (const float*)d_in[5];
    const float* bk  = (const float*)d_in[6];
    const float* Wv  = (const float*)d_in[7];
    const float* bv  = (const float*)d_in[8];
    const float* Wo  = (const float*)d_in[9];
    const float* bo  = (const float*)d_in[10];

    const size_t NX = (size_t)BATCH * S_LEN * D_MODEL;  // 4,194,304
    const size_t NW = (size_t)D_MODEL * D_MODEL;        // 1,048,576

    // Workspace layout (bf16 halves): ~64 MB total
    unsigned short* ws  = (unsigned short*)d_ws;
    size_t off = 0;
    auto take = [&](size_t n) { unsigned short* p = ws + off; off += n; return p; };
    unsigned short* xq_bf = take(NX);
    unsigned short* xk_bf = take(NX);
    unsigned short* xv_bf = take(NX);
    unsigned short* Wq_bf = take(NW);
    unsigned short* Wk_bf = take(NW);
    unsigned short* Wv_bf = take(NW);
    unsigned short* Wo_bf = take(NW);
    unsigned short* Qh    = take(NX);   // [B][H][S][dk]
    unsigned short* Kh    = take(NX);   // [B][H][S][dk]
    unsigned short* Vt    = take(NX);   // [B][H][dk][S]
    unsigned short* ctxb  = take(NX);   // [B][S][D]

    // 1) conversions (vectorized x4)
    {
        const int T = 256;
        const int nx4 = (int)(NX / 4), nw4 = (int)(NW / 4);
        cvt_bf16x4_kernel<<<(nx4 + T - 1) / T, T, 0, stream>>>((const float4*)x_q, (uint2*)xq_bf, nx4);
        cvt_bf16x4_kernel<<<(nx4 + T - 1) / T, T, 0, stream>>>((const float4*)x_k, (uint2*)xk_bf, nx4);
        cvt_bf16x4_kernel<<<(nx4 + T - 1) / T, T, 0, stream>>>((const float4*)x_v, (uint2*)xv_bf, nx4);
        cvt_bf16x4_kernel<<<(nw4 + T - 1) / T, T, 0, stream>>>((const float4*)Wq, (uint2*)Wq_bf, nw4);
        cvt_bf16x4_kernel<<<(nw4 + T - 1) / T, T, 0, stream>>>((const float4*)Wk, (uint2*)Wk_bf, nw4);
        cvt_bf16x4_kernel<<<(nw4 + T - 1) / T, T, 0, stream>>>((const float4*)Wv, (uint2*)Wv_bf, nw4);
        cvt_bf16x4_kernel<<<(nw4 + T - 1) / T, T, 0, stream>>>((const float4*)Wo, (uint2*)Wo_bf, nw4);
    }

    // 2) projections (16 rows x 64 cols per wave)
    dim3 gemm_grid((BATCH * S_LEN) / 16, D_MODEL / 64);
    proj_qkv_kernel<<<gemm_grid, 32, 0, stream>>>(xq_bf, Wq_bf, bq, Qh, 0);
    proj_qkv_kernel<<<gemm_grid, 32, 0, stream>>>(xk_bf, Wk_bf, bk, Kh, 0);
    proj_qkv_kernel<<<gemm_grid, 32, 0, stream>>>(xv_bf, Wv_bf, bv, Vt, 1);

    // 3) fused attention (+ p_attn streaming write), 4 waves / workgroup
    float* out_p    = (float*)d_out;
    float* p_attn_p = out_p + NX;
    dim3 attn_grid(S_LEN / 16, NHEAD, BATCH);
    attn_kernel<<<attn_grid, 128, 16 * PROW * sizeof(float), stream>>>(Qh, Kh, Vt,
                                                                       p_attn_p, ctxb);

    // 4) output projection
    proj_out_kernel<<<gemm_grid, 32, 0, stream>>>(ctxb, Wo_bf, bo, out_p);
}
